// Model_22093311771138
// MI455X (gfx1250) — matrix operand order repaired and checked by reading, not compiled, not run on hardware
//
#include <hip/hip_runtime.h>

// Problem dims (fixed by reference)
#define B_TOTAL 64
#define S_DIM   784
#define D_DIM   4096
#define C_DIM   10
#define L_DIM   100
#define D_HALF  (D_DIM / 2)   // 2048 h2-pairs per row

// Encode tiling
#define DT_H2    128   // h2 (2-d) elements per tile per thread -> 256 d's per WG
#define B_PER_WG 4     // batches sharing each id_weight load

typedef _Float16 h2   __attribute__((ext_vector_type(2)));
typedef _Float16 h8   __attribute__((ext_vector_type(8)));
typedef _Float16 v16h __attribute__((ext_vector_type(16)));
typedef float    v8f  __attribute__((ext_vector_type(8)));
typedef int      i4   __attribute__((ext_vector_type(4)));

// Pointer flavors for the async global->LDS builtin (param0 printed by clang
// as "int __vector_size__(16) __device__ *", i.e. non-const v4i in AS1).
typedef __attribute__((address_space(1))) i4* gas_b128_t;
typedef __attribute__((address_space(3))) i4* las_b128_t;

// gfx1250 async global->LDS copy (ASYNCcnt-tracked), guarded: falls back to
// plain vector copies if this toolchain does not expose the builtins.
#if defined(__has_builtin)
#  if __has_builtin(__builtin_amdgcn_global_load_async_to_lds_b128) && \
      __has_builtin(__builtin_amdgcn_s_wait_asynccnt)
#    define USE_ASYNC_LDS 1
#  endif
#endif
#ifndef USE_ASYNC_LDS
#  define USE_ASYNC_LDS 0
#endif

// ---------------------------------------------------------------------------
// Kernel 1: one-time conversions into workspace.
//   wf16 : classify_weight [10,4096] f32 -> padded [16,4096] f16
//   vtf16: value_table     [100,4096] f32 -> f16 (values are exactly +-1)
//   idf16: id_weight       [784,4096] f32 -> f16 (values are exactly +-1)
// ---------------------------------------------------------------------------
__global__ __launch_bounds__(256) void hdc_prep(
        const float* __restrict__ cw, const float* __restrict__ vt,
        const float* __restrict__ idw,
        _Float16* __restrict__ wf16, _Float16* __restrict__ vtf16,
        _Float16* __restrict__ idf16) {
    int g      = blockIdx.x * 256 + threadIdx.x;
    int stride = gridDim.x * 256;
    for (int i = g; i < 16 * D_DIM; i += stride) {
        int row = i >> 12;
        wf16[i] = (row < C_DIM) ? (_Float16)cw[(size_t)row * D_DIM + (i & (D_DIM - 1))]
                                : (_Float16)0.0f;
    }
    for (int i = g; i < L_DIM * D_DIM; i += stride) vtf16[i] = (_Float16)vt[i];
    for (int i = g; i < S_DIM * D_DIM; i += stride) idf16[i] = (_Float16)idw[i];
}

// ---------------------------------------------------------------------------
// Kernel 2: encode.  Grid = (16 d-tiles, 16 batch-groups), 128 threads.
//   - value_table d-slice staged in LDS as packed f16 (100 x 128 h2 = 50 KB),
//     via async global->LDS b128 copies when available (ASYNCcnt path)
//   - level indices quantized once per batch into LDS as u8
//   - each packed id load feeds 4 batch accumulators via v_pk_fma_f16
//   - f16 accumulation is EXACT here: partial sums are integers, |sum|<=784
// ---------------------------------------------------------------------------
__global__ __launch_bounds__(128) void hdc_encode(
        const float*    __restrict__ x,
        const _Float16* __restrict__ idf16,
        const _Float16* __restrict__ vtf16,
        _Float16*       __restrict__ enc) {
    __shared__ h2            vt_l[L_DIM * DT_H2];       // 50 KB
    __shared__ unsigned char idxs[B_PER_WG][S_DIM];     // ~3 KB

    const int tid = threadIdx.x;   // 0..127
    const int dt  = blockIdx.x;    // 0..15
    const int bg  = blockIdx.y;    // 0..15
    const int d2  = dt * DT_H2 + tid;   // h2 column index, 0..2047
    const int b0  = bg * B_PER_WG;

    // --- stage value_table tile: 100 rows x 512 B = 51200 B, 16 B/lane/step ---
    {
        const char* gbase = (const char*)vtf16;
        char*       lbase = (char*)vt_l;
        #pragma unroll 5
        for (int step = 0; step < 25; ++step) {
            int boff = step * (128 * 16) + tid * 16;      // byte offset in tile
            int row  = boff >> 9;                         // /512 (tile row bytes)
            int colb = boff & 511;
            const char* gsrc = gbase + (size_t)row * (D_DIM * 2) + dt * (DT_H2 * 4) + colb;
            char*       ldst = lbase + boff;
#if USE_ASYNC_LDS
            __builtin_amdgcn_global_load_async_to_lds_b128(
                (gas_b128_t)gsrc, (las_b128_t)ldst, 0, 0);
#else
            *(uint4*)ldst = *(const uint4*)gsrc;
#endif
        }
    }

    // --- quantize x -> level index (round-half-even like jnp.round), as u8 ---
    for (int i = tid; i < B_PER_WG * S_DIM; i += 128) {
        int bb = i / S_DIM;
        int s  = i - bb * S_DIM;
        float xv = x[(size_t)(b0 + bb) * S_DIM + s];
        int li = (int)rintf(xv * (float)(L_DIM - 1));
        li = li < 0 ? 0 : (li > L_DIM - 1 ? L_DIM - 1 : li);
        idxs[bb][s] = (unsigned char)li;
    }

#if USE_ASYNC_LDS
    __builtin_amdgcn_s_wait_asynccnt(0);
#endif
    __syncthreads();

    const h2* idp = (const h2*)idf16;   // [784][2048]
    h2 a0 = {0, 0}, a1 = {0, 0}, a2 = {0, 0}, a3 = {0, 0};
    for (int s0 = 0; s0 < S_DIM; s0 += 4) {               // 784 = 4*196
        int pf = (s0 + 8 < S_DIM) ? (s0 + 8) : (S_DIM - 1);
        __builtin_prefetch(&idp[(size_t)pf * D_HALF + d2], 0, 0);
        #pragma unroll
        for (int u = 0; u < 4; ++u) {
            int s = s0 + u;
            h2 idv = idp[(size_t)s * D_HALF + d2];
            a0 += idv * vt_l[(int)idxs[0][s] * DT_H2 + tid];
            a1 += idv * vt_l[(int)idxs[1][s] * DT_H2 + tid];
            a2 += idv * vt_l[(int)idxs[2][s] * DT_H2 + tid];
            a3 += idv * vt_l[(int)idxs[3][s] * DT_H2 + tid];
        }
    }

    // hard_quantize (exact integer sums) -> packed +-1 f16
    h2* encp = (h2*)enc;
    h2 r;
    r[0] = (float)a0[0] > 0.f ? (_Float16)1.f : (_Float16)-1.f;
    r[1] = (float)a0[1] > 0.f ? (_Float16)1.f : (_Float16)-1.f;
    encp[(size_t)(b0 + 0) * D_HALF + d2] = r;
    r[0] = (float)a1[0] > 0.f ? (_Float16)1.f : (_Float16)-1.f;
    r[1] = (float)a1[1] > 0.f ? (_Float16)1.f : (_Float16)-1.f;
    encp[(size_t)(b0 + 1) * D_HALF + d2] = r;
    r[0] = (float)a2[0] > 0.f ? (_Float16)1.f : (_Float16)-1.f;
    r[1] = (float)a2[1] > 0.f ? (_Float16)1.f : (_Float16)-1.f;
    encp[(size_t)(b0 + 2) * D_HALF + d2] = r;
    r[0] = (float)a3[0] > 0.f ? (_Float16)1.f : (_Float16)-1.f;
    r[1] = (float)a3[1] > 0.f ? (_Float16)1.f : (_Float16)-1.f;
    encp[(size_t)(b0 + 3) * D_HALF + d2] = r;
}

// ---------------------------------------------------------------------------
// Kernel 3: classify via v_wmma_f32_16x16x32_f16.
//   Grid = 4 blocks (one 16-row M-tile each), 32 threads = 1 wave.
//   A frag (16x32 f16): lanes<16 row=lane, K{0..7,16..23}; lanes>=16 K{8..15,24..31}
//   B frag (32x16 f16): lanes<16 col=lane, K{0..7,16..23}; lanes>=16 K{8..15,24..31}
//   D frag (16x16 f32): lanes<16 M=r;  lanes>=16 M=8+r;  N = lane&15
// ---------------------------------------------------------------------------
__global__ __launch_bounds__(32) void hdc_classify(
        const _Float16* __restrict__ enc,
        const _Float16* __restrict__ wf16,
        float* __restrict__ out) {
    const int lane = threadIdx.x;        // 0..31
    const int n    = lane & 15;          // class column / A row-in-tile
    const int hi   = lane >> 4;          // lane group
    const int m0   = blockIdx.x * 16;    // batch-row tile base
    const int koff = hi * 8;

    const _Float16* arow = enc  + (size_t)(m0 + n) * D_DIM;
    const _Float16* bcol = wf16 + (size_t)n * D_DIM;

    v8f acc = {};
    for (int k0 = 0; k0 < D_DIM; k0 += 32) {
        h8 alo = *(const h8*)(arow + k0 + koff);        // K: koff .. koff+7
        h8 ahi = *(const h8*)(arow + k0 + koff + 16);   // K: koff+16 .. koff+23
        h8 blo = *(const h8*)(bcol + k0 + koff);        // K: koff .. koff+7
        h8 bhi = *(const h8*)(bcol + k0 + koff + 16);   // K: koff+16 .. koff+23
        v16h b = __builtin_shufflevector(blo, bhi, 0, 1, 2, 3, 4, 5, 6, 7, 8, 9, 10, 11, 12, 13, 14, 15);
        v16h a;
        #pragma unroll
        for (int i = 0; i < 8; ++i) { a[i] = alo[i]; a[8 + i] = ahi[i]; }
        acc = __builtin_amdgcn_wmma_f32_16x16x32_f16(
                  /*neg_a=*/false, a, /*neg_b=*/false, b,
                  /*c_mod=*/(short)0, acc, /*reuse_a=*/false, /*reuse_b=*/false);
    }

    if (n < C_DIM) {
        #pragma unroll
        for (int r = 0; r < 8; ++r) {
            int m = hi * 8 + r;
            out[(size_t)(m0 + m) * C_DIM + n] = acc[r];
        }
    }
}

// ---------------------------------------------------------------------------
extern "C" void kernel_launch(void* const* d_in, const int* in_sizes, int n_in,
                              void* d_out, int out_size, void* d_ws, size_t ws_size,
                              hipStream_t stream) {
    const float* x   = (const float*)d_in[0];   // [64, 784]
    const float* idw = (const float*)d_in[1];   // [784, 4096]
    const float* vt  = (const float*)d_in[2];   // [100, 4096]
    const float* cw  = (const float*)d_in[3];   // [10, 4096]
    float* out = (float*)d_out;                 // [64, 10]

    // Workspace layout (f16 units): enc | wf16 | vtf16 | idf16  (~7.9 MB total)
    _Float16* enc   = (_Float16*)d_ws;
    _Float16* wf16  = enc   + (size_t)B_TOTAL * D_DIM;
    _Float16* vtf16 = wf16  + (size_t)16 * D_DIM;
    _Float16* idf16 = vtf16 + (size_t)L_DIM * D_DIM;

    hdc_prep<<<dim3(2048), dim3(256), 0, stream>>>(cw, vt, idw, wf16, vtf16, idf16);
    hdc_encode<<<dim3(D_HALF / DT_H2, B_TOTAL / B_PER_WG), dim3(128), 0, stream>>>(
        x, idf16, vtf16, enc);
    hdc_classify<<<dim3(B_TOTAL / 16), dim3(32), 0, stream>>>(enc, wf16, out);
}
